// TransformerBlock_67886253080750
// MI455X (gfx1250) — compile-verified, matrix operand order checked by reading, not run on hardware
//
#include <hip/hip_runtime.h>
#include <hip/hip_bf16.h>

// ---------------------------------------------------------------------------
// Transformer block for MI455X (gfx1250): bf16 WMMA GEMMs + flash attention,
// async global->LDS copies, mask-tile skip.
// ---------------------------------------------------------------------------

#define BBATCH 2
#define CCH    512
#define SSEQ   2048
#define NH     8
#define HD     64
#define MROWS  (BBATCH * SSEQ)   // 4096

#define USE_ASYNC 1

typedef unsigned short u16;
typedef unsigned int   u32;

typedef __attribute__((ext_vector_type(16))) __bf16 v16bf;
typedef __attribute__((ext_vector_type(8)))  float  v8f;

struct AFrag {
  union { v16bf v; u32 w[8]; };
};

__device__ inline u16 f2bf(float f) {
  u32 u = __float_as_uint(f);
  u = (u + 0x7FFFu + ((u >> 16) & 1u)) >> 16;   // round-to-nearest-even
  return (u16)u;
}

// ---- CDNA5 async global->LDS copy (ASYNCcnt-tracked) ----------------------
__device__ inline void async_ld_b128(void* lds, const void* g) {
#if USE_ASYNC
  const u32 loff = (u32)(uintptr_t)lds;                 // LDS offset = addr[31:0]
  const unsigned long long ga = (unsigned long long)(uintptr_t)g;
  asm volatile("global_load_async_to_lds_b128 %0, %1, off"
               :: "v"(loff), "v"(ga) : "memory");
#else
  *(uint4*)lds = *(const uint4*)g;
#endif
}

__device__ inline void wait_async0() {
#if USE_ASYNC
#if defined(__has_builtin)
#if __has_builtin(__builtin_amdgcn_s_wait_asynccnt)
  __builtin_amdgcn_s_wait_asynccnt(0);
#else
  asm volatile("s_wait_asynccnt 0" ::: "memory");
#endif
#else
  asm volatile("s_wait_asynccnt 0" ::: "memory");
#endif
#endif
}

// Build a 16x32-bf16 WMMA fragment (A or B side) from an LDS row contiguous
// in K.  Per ISA 7.12.2: lane half selects K {0..7,16..23} vs {8..15,24..31};
// VGPR pairs are consecutive K -> each fragment dword is one aligned b32 read
// (compiler folds the 8 reads into two ds_load_b128).
__device__ inline void load_frag(const u16* base, int half, int kc, AFrag& f) {
#pragma unroll
  for (int p = 0; p < 8; ++p) {
    const int kk = ((p < 4) ? (2 * p) : (16 + 2 * (p - 4))) + 8 * half + kc;
    f.w[p] = *(const u32*)(base + kk);
  }
}

// ---------------------------------------------------------------------------
// fp32 -> bf16 conversion
// ---------------------------------------------------------------------------
__global__ __launch_bounds__(256) void cvt_f32_bf16(const float* __restrict__ in,
                                                    u16* __restrict__ out, int n) {
  int i = blockIdx.x * 256 + threadIdx.x;
  if (i < n) out[i] = f2bf(in[i]);
}

// ---------------------------------------------------------------------------
// Channels-first LayerNorm over C with transpose:
//   in (B, C, S) fp32  ->  out (B*S, C) bf16
// ---------------------------------------------------------------------------
__global__ __launch_bounds__(256) void ln_cf_bf16(const float* __restrict__ x,
                                                  const float* __restrict__ w,
                                                  const float* __restrict__ bias,
                                                  u16* __restrict__ out) {
  const int bs = blockIdx.x;
  const int b = bs >> 11;            // / SSEQ
  const int s = bs & (SSEQ - 1);
  const int t = threadIdx.x;
  const float* xb = x + (size_t)b * CCH * SSEQ + s;

  const float v0 = xb[(size_t)t * SSEQ];
  const float v1 = xb[(size_t)(t + 256) * SSEQ];

  __shared__ float r1[256], r2[256];
  r1[t] = v0 + v1;
  r2[t] = v0 * v0 + v1 * v1;
  __syncthreads();
#pragma unroll
  for (int o = 128; o > 0; o >>= 1) {
    if (t < o) { r1[t] += r1[t + o]; r2[t] += r2[t + o]; }
    __syncthreads();
  }
  const float u   = r1[0] * (1.0f / CCH);
  const float var = r2[0] * (1.0f / CCH) - u * u;
  const float rs  = rsqrtf(var + 1e-6f);

  u16* ob = out + (size_t)bs * CCH;
  ob[t]       = f2bf((v0 - u) * rs * w[t]       + bias[t]);
  ob[t + 256] = f2bf((v1 - u) * rs * w[t + 256] + bias[t + 256]);
}

// ---------------------------------------------------------------------------
// bf16 WMMA GEMM:  out(M,N) = A(M,K) @ Bw(K,N) + bias, fp32 accumulate.
// 256 threads (8 waves), 128x128 block tile, K-step 64 (2 WMMA K-chunks per
// LDS fill).  Wave grid 2x4; each wave owns 64x32 = 4x2 WMMA tiles.
// EPI 0: store bf16 row-major.   EPI 1: exact GELU, store bf16.
// EPI 2: + resid ((B,C,S) fp32), store fp32 transposed to (B,C,S).
// ---------------------------------------------------------------------------
template <int EPI>
__global__ __launch_bounds__(256)
void gemm_bf16(const u16* __restrict__ A, const u16* __restrict__ Bw,
               const float* __restrict__ bias, u16* __restrict__ outb,
               float* __restrict__ outf, const float* __restrict__ resid,
               int M, int N, int K) {
  __shared__ u16 As[128][72];    // A tile, row-major (K contiguous)
  __shared__ u16 Bst[128][72];   // B tile transposed (n rows, K contiguous)

  const int tid  = threadIdx.x;
  const int wid  = tid >> 5, lane = tid & 31;
  const int half = lane >> 4, ln16 = lane & 15;
  const int wm   = wid >> 2,  wn   = wid & 3;
  const int bm   = blockIdx.y * 128, bn = blockIdx.x * 128;

  v8f acc[4][2];
#pragma unroll
  for (int mi = 0; mi < 4; ++mi)
#pragma unroll
    for (int ni = 0; ni < 2; ++ni)
#pragma unroll
      for (int r = 0; r < 8; ++r) acc[mi][ni][r] = 0.0f;

  for (int k0 = 0; k0 < K; k0 += 64) {
    {   // A tile: 128 rows x 64 cols, async straight copy
      const int r = tid >> 1, c0 = (tid & 1) * 32;
      const u16* src = A + (size_t)(bm + r) * K + k0 + c0;
#pragma unroll
      for (int j = 0; j < 4; ++j) async_ld_b128(&As[r][c0 + j * 8], src + j * 8);
      if (k0 + 64 < K) __builtin_prefetch(src + 64, 0, 1);
    }
    {   // B tile: 64 k-rows x 128 cols, transposed store
      const int r = tid >> 2, c0 = (tid & 3) * 32;
      const u16* src = Bw + (size_t)(k0 + r) * N + bn + c0;
#pragma unroll
      for (int j = 0; j < 4; ++j) {
        const uint4 d = *(const uint4*)(src + j * 8);
        u16 tmp[8];
        *(uint4*)tmp = d;
#pragma unroll
        for (int e = 0; e < 8; ++e) Bst[c0 + j * 8 + e][r] = tmp[e];
      }
      if (k0 + 64 < K)
        __builtin_prefetch(Bw + (size_t)(k0 + 64 + r) * N + bn + c0, 0, 1);
    }
    wait_async0();
    __syncthreads();

#pragma unroll
    for (int kc = 0; kc < 64; kc += 32) {
      AFrag af[4], bf[2];
#pragma unroll
      for (int mi = 0; mi < 4; ++mi)
        load_frag(&As[wm * 64 + mi * 16 + ln16][0], half, kc, af[mi]);
#pragma unroll
      for (int ni = 0; ni < 2; ++ni)
        load_frag(&Bst[wn * 32 + ni * 16 + ln16][0], half, kc, bf[ni]);
#pragma unroll
      for (int mi = 0; mi < 4; ++mi)
#pragma unroll
        for (int ni = 0; ni < 2; ++ni)
          acc[mi][ni] = __builtin_amdgcn_wmma_f32_16x16x32_bf16(
              false, af[mi].v, false, bf[ni].v, (short)0, acc[mi][ni], false, false);
    }
    __syncthreads();
  }

#pragma unroll
  for (int mi = 0; mi < 4; ++mi) {
#pragma unroll
    for (int ni = 0; ni < 2; ++ni) {
      const int col = bn + wn * 32 + ni * 16 + ln16;
      const float bv = bias[col];
#pragma unroll
      for (int r = 0; r < 8; ++r) {
        const int row = bm + wm * 64 + mi * 16 + r + 8 * half;
        float val = acc[mi][ni][r] + bv;
        if (EPI == 0) {
          outb[(size_t)row * N + col] = f2bf(val);
        } else if (EPI == 1) {
          val = 0.5f * val * (1.0f + erff(val * 0.70710678118f));
          outb[(size_t)row * N + col] = f2bf(val);
        } else {
          const int b = row >> 11;               // row / SSEQ
          const int s = row & (SSEQ - 1);
          const size_t idx = (size_t)b * CCH * SSEQ + (size_t)col * SSEQ + s;
          outf[idx] = val + resid[idx];
        }
      }
    }
  }
}

// ---------------------------------------------------------------------------
// Flash attention. Block = 128 threads (4 waves), one (b, h, 64 q-rows).
// Scores & O live in WMMA accumulators; online softmax with half-wave
// shuffles; mask tile staged to LDS with a double-buffered "any masked"
// flag so clean tiles skip per-element masking entirely.
// ---------------------------------------------------------------------------
__global__ __launch_bounds__(128)
void attn_kernel(const u16* __restrict__ Q, const u16* __restrict__ Kmat,
                 const u16* __restrict__ V, const unsigned char* __restrict__ mask,
                 u16* __restrict__ ctx) {
  __shared__ u16 Qs[64][72];       // q-rows, d contiguous
  __shared__ u16 Ks[64][72];       // key-rows, d contiguous (B-frag Q@K^T)
  __shared__ u16 Vt[64][72];       // d-rows, key contiguous (B-frag P@V)
  __shared__ u16 Ps[4][16][72];    // per-wave P staging, key contiguous
  __shared__ unsigned char Ms[64][64];
  __shared__ int tmask[2];

  const int tid  = threadIdx.x;
  const int wid  = tid >> 5, lane = tid & 31;
  const int half = lane >> 4, ln16 = lane & 15;
  const int q0 = blockIdx.x * 64;
  const int h  = blockIdx.y;
  const int b  = blockIdx.z;

  {
    const int r = tid >> 1, c0 = (tid & 1) * 32;
    const u16* src = Q + ((size_t)(b * SSEQ + q0 + r)) * CCH + h * HD + c0;
#pragma unroll
    for (int j = 0; j < 4; ++j)
      async_ld_b128(&Qs[r][c0 + j * 8], src + j * 8);
  }
  if (tid < 2) tmask[tid] = 0;

  float m_r[8], l_r[8];
  v8f accO[4];
#pragma unroll
  for (int r = 0; r < 8; ++r) { m_r[r] = -1e30f; l_r[r] = 0.0f; }
#pragma unroll
  for (int nt = 0; nt < 4; ++nt)
#pragma unroll
    for (int r = 0; r < 8; ++r) accO[nt][r] = 0.0f;

  wait_async0();
  __syncthreads();

  const size_t mbase = (size_t)b * SSEQ * SSEQ;

  for (int kv0 = 0; kv0 < SSEQ; kv0 += 64) {
    const int par = (kv0 >> 6) & 1;
    {
      const int r = tid >> 1, c0 = (tid & 1) * 32;
      // K tile: async straight copy
      const u16* ksrc = Kmat + ((size_t)(b * SSEQ + kv0 + r)) * CCH + h * HD + c0;
#pragma unroll
      for (int j = 0; j < 4; ++j)
        async_ld_b128(&Ks[r][c0 + j * 8], ksrc + j * 8);
      // V tile: transposed store
      const u16* vsrc = V + ((size_t)(b * SSEQ + kv0 + r)) * CCH + h * HD + c0;
#pragma unroll
      for (int j = 0; j < 4; ++j) {
        const uint4 d = *(const uint4*)(vsrc + j * 8);
        u16 tmp[8];
        *(uint4*)tmp = d;
#pragma unroll
        for (int e = 0; e < 8; ++e) Vt[c0 + j * 8 + e][r] = tmp[e];
      }
      // mask tile: stage + any-set flag
      const unsigned char* msrc = mask + mbase + (size_t)(q0 + r) * SSEQ + kv0 + c0;
      const uint4 m0 = *(const uint4*)(msrc);
      const uint4 m1 = *(const uint4*)(msrc + 16);
      *(uint4*)&Ms[r][c0]      = m0;
      *(uint4*)&Ms[r][c0 + 16] = m1;
      const u32 any = m0.x | m0.y | m0.z | m0.w | m1.x | m1.y | m1.z | m1.w;
      if (any) atomicOr(&tmask[par], 1);
    }
    wait_async0();
    __syncthreads();

    // S = (Q @ K^T) / sqrt(D)
    v8f sacc[4];
#pragma unroll
    for (int nt = 0; nt < 4; ++nt)
#pragma unroll
      for (int r = 0; r < 8; ++r) sacc[nt][r] = 0.0f;
#pragma unroll
    for (int nt = 0; nt < 4; ++nt) {
#pragma unroll
      for (int kc = 0; kc < 64; kc += 32) {
        AFrag a, bb;
        load_frag(&Qs[wid * 16 + ln16][0], half, kc, a);
        load_frag(&Ks[nt * 16 + ln16][0], half, kc, bb);
        sacc[nt] = __builtin_amdgcn_wmma_f32_16x16x32_bf16(
            false, a.v, false, bb.v, (short)0, sacc[nt], false, false);
      }
    }

    // scale (+ masking only when the tile has any masked element)
    if (tmask[par]) {
#pragma unroll
      for (int nt = 0; nt < 4; ++nt)
#pragma unroll
        for (int r = 0; r < 8; ++r) {
          float sv = sacc[nt][r] * 0.125f;
          const int rr = wid * 16 + r + 8 * half;
          if (Ms[rr][nt * 16 + ln16]) sv = -1e9f;
          sacc[nt][r] = sv;
        }
    } else {
#pragma unroll
      for (int nt = 0; nt < 4; ++nt)
#pragma unroll
        for (int r = 0; r < 8; ++r) sacc[nt][r] *= 0.125f;
    }

    // online softmax: each lane's 8 acc rows == its 8 stat slots.
    float mloc[8];
#pragma unroll
    for (int r = 0; r < 8; ++r) {
      float mv = sacc[0][r];
#pragma unroll
      for (int nt = 1; nt < 4; ++nt) mv = fmaxf(mv, sacc[nt][r]);
      mloc[r] = mv;
    }
#pragma unroll
    for (int d = 1; d < 16; d <<= 1)
#pragma unroll
      for (int r = 0; r < 8; ++r)
        mloc[r] = fmaxf(mloc[r], __shfl_xor(mloc[r], d, 32));

    float alpha[8], sl[8];
#pragma unroll
    for (int r = 0; r < 8; ++r) {
      const float mn = fmaxf(m_r[r], mloc[r]);
      alpha[r] = __expf(m_r[r] - mn);
      m_r[r] = mn;
      sl[r] = 0.0f;
    }
#pragma unroll
    for (int nt = 0; nt < 4; ++nt)
#pragma unroll
      for (int r = 0; r < 8; ++r) {
        const float p = __expf(sacc[nt][r] - m_r[r]);
        sacc[nt][r] = p;
        sl[r] += p;
      }
#pragma unroll
    for (int d = 1; d < 16; d <<= 1)
#pragma unroll
      for (int r = 0; r < 8; ++r) sl[r] += __shfl_xor(sl[r], d, 32);
#pragma unroll
    for (int r = 0; r < 8; ++r) l_r[r] = l_r[r] * alpha[r] + sl[r];
#pragma unroll
    for (int nt = 0; nt < 4; ++nt)
#pragma unroll
      for (int r = 0; r < 8; ++r) accO[nt][r] *= alpha[r];

    // stage P (bf16) per wave; C-layout -> A-layout via LDS
#pragma unroll
    for (int nt = 0; nt < 4; ++nt)
#pragma unroll
      for (int r = 0; r < 8; ++r)
        Ps[wid][r + 8 * half][nt * 16 + ln16] = f2bf(sacc[nt][r]);

    // O += P @ V
#pragma unroll
    for (int nt = 0; nt < 4; ++nt) {
#pragma unroll
      for (int kc = 0; kc < 64; kc += 32) {
        AFrag a, bb;
        load_frag(&Ps[wid][ln16][0], half, kc, a);
        load_frag(&Vt[nt * 16 + ln16][0], half, kc, bb);
        accO[nt] = __builtin_amdgcn_wmma_f32_16x16x32_bf16(
            false, a.v, false, bb.v, (short)0, accO[nt], false, false);
      }
    }
    __syncthreads();
    if (tid == 0) tmask[par] = 0;   // other parity slot used next iter
  }

#pragma unroll
  for (int nt = 0; nt < 4; ++nt)
#pragma unroll
    for (int r = 0; r < 8; ++r) {
      const float o = accO[nt][r] / l_r[r];
      const int qr = q0 + wid * 16 + r + 8 * half;
      ctx[((size_t)(b * SSEQ + qr)) * CCH + h * HD + nt * 16 + ln16] = f2bf(o);
    }
}

// ---------------------------------------------------------------------------
// Host-side orchestration
// ---------------------------------------------------------------------------
static constexpr size_t SZ_W   = (size_t)CCH * CCH * 2;
static constexpr size_t SZ_W1  = (size_t)CCH * 4 * CCH * 2;
static constexpr size_t SZ_ACT = (size_t)MROWS * CCH * 2;
static constexpr size_t SZ_Z1  = (size_t)BBATCH * CCH * SSEQ * 4;

static constexpr size_t OFF_WQ  = 0;
static constexpr size_t OFF_WK  = OFF_WQ + SZ_W;
static constexpr size_t OFF_WV  = OFF_WK + SZ_W;
static constexpr size_t OFF_WO  = OFF_WV + SZ_W;
static constexpr size_t OFF_W1  = OFF_WO + SZ_W;
static constexpr size_t OFF_W2  = OFF_W1 + SZ_W1;
static constexpr size_t OFF_ZT  = OFF_W2 + SZ_W1;
static constexpr size_t OFF_Q   = OFF_ZT + SZ_ACT;
static constexpr size_t OFF_K   = OFF_Q + SZ_ACT;
static constexpr size_t OFF_V   = OFF_K + SZ_ACT;
static constexpr size_t OFF_CTX = OFF_V + SZ_ACT;
static constexpr size_t OFF_Z2T = OFF_CTX + SZ_ACT;
static constexpr size_t OFF_Z1  = OFF_Z2T + SZ_ACT;
static constexpr size_t OFF_H   = OFF_Z1 + SZ_Z1;

extern "C" void kernel_launch(void* const* d_in, const int* in_sizes, int n_in,
                              void* d_out, int out_size, void* d_ws, size_t ws_size,
                              hipStream_t stream) {
  (void)in_sizes; (void)n_in; (void)out_size; (void)ws_size;
  const float* x    = (const float*)d_in[0];
  const unsigned char* mask = (const unsigned char*)d_in[1];
  const float* ln_w = (const float*)d_in[2];
  const float* ln_b = (const float*)d_in[3];
  const float* wq = (const float*)d_in[4];
  const float* bq = (const float*)d_in[5];
  const float* wk = (const float*)d_in[6];
  const float* bk = (const float*)d_in[7];
  const float* wv = (const float*)d_in[8];
  const float* bv = (const float*)d_in[9];
  const float* wo = (const float*)d_in[10];
  const float* bo = (const float*)d_in[11];
  const float* w1 = (const float*)d_in[12];
  const float* b1 = (const float*)d_in[13];
  const float* w2 = (const float*)d_in[14];
  const float* b2 = (const float*)d_in[15];

  char* ws = (char*)d_ws;
  u16* wq_bf = (u16*)(ws + OFF_WQ);
  u16* wk_bf = (u16*)(ws + OFF_WK);
  u16* wv_bf = (u16*)(ws + OFF_WV);
  u16* wo_bf = (u16*)(ws + OFF_WO);
  u16* w1_bf = (u16*)(ws + OFF_W1);
  u16* w2_bf = (u16*)(ws + OFF_W2);
  u16* zt    = (u16*)(ws + OFF_ZT);
  u16* qb    = (u16*)(ws + OFF_Q);
  u16* kb    = (u16*)(ws + OFF_K);
  u16* vb    = (u16*)(ws + OFF_V);
  u16* ctxb  = (u16*)(ws + OFF_CTX);
  u16* z2t   = (u16*)(ws + OFF_Z2T);
  float* z1f = (float*)(ws + OFF_Z1);
  u16* hb    = (u16*)(ws + OFF_H);

  const int nW  = CCH * CCH;
  const int nW1 = CCH * 4 * CCH;
  cvt_f32_bf16<<<(nW + 255) / 256, 256, 0, stream>>>(wq, wq_bf, nW);
  cvt_f32_bf16<<<(nW + 255) / 256, 256, 0, stream>>>(wk, wk_bf, nW);
  cvt_f32_bf16<<<(nW + 255) / 256, 256, 0, stream>>>(wv, wv_bf, nW);
  cvt_f32_bf16<<<(nW + 255) / 256, 256, 0, stream>>>(wo, wo_bf, nW);
  cvt_f32_bf16<<<(nW1 + 255) / 256, 256, 0, stream>>>(w1, w1_bf, nW1);
  cvt_f32_bf16<<<(nW1 + 255) / 256, 256, 0, stream>>>(w2, w2_bf, nW1);

  // LN(x) transposed to (B*S, C) bf16
  ln_cf_bf16<<<MROWS, 256, 0, stream>>>(x, ln_w, ln_b, zt);

  // Q, K, V projections
  dim3 g512(512 / 128, MROWS / 128);
  gemm_bf16<0><<<g512, 256, 0, stream>>>(zt, wq_bf, bq, qb, nullptr, nullptr, MROWS, CCH, CCH);
  gemm_bf16<0><<<g512, 256, 0, stream>>>(zt, wk_bf, bk, kb, nullptr, nullptr, MROWS, CCH, CCH);
  gemm_bf16<0><<<g512, 256, 0, stream>>>(zt, wv_bf, bv, vb, nullptr, nullptr, MROWS, CCH, CCH);

  // flash attention -> ctx (B*S, C) bf16
  attn_kernel<<<dim3(SSEQ / 64, NH, BBATCH), 128, 0, stream>>>(qb, kb, vb, mask, ctxb);

  // z1 = ctx @ Wo + bo (+ x residual, stored (B,C,S) fp32)
  gemm_bf16<2><<<g512, 256, 0, stream>>>(ctxb, wo_bf, bo, nullptr, z1f, x, MROWS, CCH, CCH);

  // z2 = LN(z1) transposed bf16
  ln_cf_bf16<<<MROWS, 256, 0, stream>>>(z1f, ln_w, ln_b, z2t);

  // h = GELU(z2 @ W1 + b1)
  dim3 g2048(2048 / 128, MROWS / 128);
  gemm_bf16<1><<<g2048, 256, 0, stream>>>(z2t, w1_bf, b1, hb, nullptr, nullptr, MROWS, 4 * CCH, CCH);

  // out = h @ W2 + b2 + z1, stored (B,C,S) fp32
  gemm_bf16<2><<<g512, 256, 0, stream>>>(hb, w2_bf, b2, nullptr, (float*)d_out, z1f, MROWS, CCH, 4 * CCH);
}
